// LSTM_custom_2800318677188
// MI455X (gfx1250) — matrix operand order repaired and checked by reading, not compiled, not run on hardware
//
#include <hip/hip_runtime.h>
#include <hip/hip_bf16.h>

typedef __attribute__((ext_vector_type(16))) _Float16 v16h;
typedef __attribute__((ext_vector_type(8)))  _Float16 v8h;
typedef __attribute__((ext_vector_type(8)))  float    v8f;

#define HSZ    64
#define T_LEN  450
#define B_TOT  2048
#define BT     16        // batch rows per workgroup (one WMMA M-tile)
#define NTH    256       // 8 wave32s -> 2 waves per SIMD32 for co-execution

// ---- dynamic LDS byte offsets (all 32B aligned) ----
#define OFF_BC1   0        // swizzled B frags, ctx LSTM1  (K=64)  : 32768 B
#define OFF_BC2   32768    // ctx LSTM2                            : 32768 B
#define OFF_BS1   65536    // stacked LSTM1 (K=128)                : 65536 B
#define OFF_BS2   131072   // stacked LSTM2 (K=128)                : 65536 B
#define OFF_BPRO  196608   // projection (K=128, N=32)             : 8192 B
#define OFF_AC1   204800   // A: h_c1 f16 [16][64]                 : 2048 B
#define OFF_AC2   206848   // A: h_c2 f16 [16][64]                 : 2048 B
#define OFF_AS1   208896   // A: [h1(64)|xraw(3)|proj(32)|pad] f16 : 4096 B
#define OFF_AS2   212992   // A: [h2(64)|h1(64)] f16 [16][128]     : 4096 B
#define OFF_APRO  217088   // A: [out1|out2] f16 [16][128]         : 4096 B
#define OFF_G     221184   // gate buf f32 [16][256]               : 16384 B
#define OFF_H2F   237568   // h2 f32 [16][64]                      : 4096 B
#define OFF_XC1   241664   // ctx1 x f32 [16][4]                   : 256 B
#define OFF_XC2   241920   // ctx2 x f32 [16]                      : 64 B
#define OFF_WIC1  241984   // Wih_c1 f32 [256][4]                  : 4096 B
#define OFF_WIC2  246080   // Wih_c2 f32 [256]                     : 1024 B
#define OFF_BI_C1 247104   // bias (bih+bhh) f32 [256]             : 1024 B
#define OFF_BI_C2 248128
#define OFF_BI_S1 249152
#define OFF_BI_S2 250176
#define OFF_BPROB 251200   // b_pro f32 [32]
#define OFF_WLIN  251328   // W_lin f32 [64]
#define OFF_BLIN  251584   // b_lin f32 [1]
#define SMEM_BYTES 251648

// A fragment (16x32 f16) in contiguous K order (B swizzle matches): lane m=lane&15, hh=lane>>4,
// elems 0..15 -> K = ko + hh*16 + i
__device__ __forceinline__ v16h load_a_frag(const _Float16* A, int stride, int m, int hh, int ko) {
  const _Float16* p = A + m * stride + ko + hh * 16;
  v8h lo = *(const v8h*)p;
  v8h hi = *(const v8h*)(p + 8);
  return __builtin_shufflevector(lo, hi, 0,1,2,3,4,5,6,7,8,9,10,11,12,13,14,15);
}

// One LSTM gate GEMM: g[16][256] = A[16][K] * B[K][256]; 2 N-tiles per wave (8 waves).
// B is pre-swizzled: frag (kt,nt) stored lane-major, 16 contiguous halfs per lane.
template<int KT>
__device__ __forceinline__ void gate_gemm(const _Float16* A, int stride,
                                          const _Float16* Bf, float* g,
                                          int wave, int lane, int hh, int m) {
  #pragma unroll
  for (int tix = 0; tix < 2; ++tix) {
    int nt = wave * 2 + tix;
    v8f acc = {};
    #pragma unroll
    for (int kt = 0; kt < KT; ++kt) {
      v16h a = load_a_frag(A, stride, m, hh, kt * 32);
      v16h b = *(const v16h*)(Bf + (((kt * 16 + nt) * 32) + lane) * 16);
      acc = __builtin_amdgcn_wmma_f32_16x16x32_f16(false, a, false, b, (short)0, acc, false, false);
    }
    #pragma unroll
    for (int r = 0; r < 8; ++r)
      g[(r + hh * 8) * 256 + nt * 16 + m] = acc[r];   // C/D layout: M=r+hh*8, N=m
  }
}

// Elementwise gate nonlinearity + state update for 4 (m,n) cells per thread.
template<int NX>
__device__ __forceinline__ void lstm_elem(const float* g, const float* bias,
    const float* WihL, const float* xrow, float* c,
    _Float16* hA0, int s0, int o0,
    _Float16* hA1, int s1, int o1,
    float* hf, int mm, int n0) {
  float xv[NX ? NX : 1];
  #pragma unroll
  for (int d = 0; d < NX; ++d) xv[d] = xrow[d];
  #pragma unroll
  for (int q = 0; q < 4; ++q) {
    int n = n0 + q;
    float gi = g[mm*256 + n]       + bias[n];
    float gf = g[mm*256 + 64 + n]  + bias[64 + n];
    float gg = g[mm*256 + 128 + n] + bias[128 + n];
    float go = g[mm*256 + 192 + n] + bias[192 + n];
    #pragma unroll
    for (int d = 0; d < NX; ++d) {
      gi += xv[d] * WihL[n * NX + d];
      gf += xv[d] * WihL[(64 + n) * NX + d];
      gg += xv[d] * WihL[(128 + n) * NX + d];
      go += xv[d] * WihL[(192 + n) * NX + d];
    }
    float ig = 1.f / (1.f + __expf(-gi));
    float fg = 1.f / (1.f + __expf(-gf));
    float gv = tanhf(gg);
    float og = 1.f / (1.f + __expf(-go));
    float cc = fg * c[q] + ig * gv;
    c[q] = cc;
    float h = og * tanhf(cc);
    hA0[mm * s0 + o0 + n] = (_Float16)h;
    if (hA1) hA1[mm * s1 + o1 + n] = (_Float16)h;
    if (hf)  hf[mm * 64 + n] = h;
  }
}

__global__ void __launch_bounds__(NTH) fused_lstm_stack_kernel(
    const float* __restrict__ inputs, const float* __restrict__ ctx1, const float* __restrict__ ctx2,
    const float* __restrict__ Wih_c1, const float* __restrict__ Whh_c1,
    const float* __restrict__ bih_c1, const float* __restrict__ bhh_c1,
    const float* __restrict__ Wih_c2, const float* __restrict__ Whh_c2,
    const float* __restrict__ bih_c2, const float* __restrict__ bhh_c2,
    const float* __restrict__ W_pro,  const float* __restrict__ b_pro,
    const float* __restrict__ Wih_1,  const float* __restrict__ Whh_1,
    const float* __restrict__ bih_1,  const float* __restrict__ bhh_1,
    const float* __restrict__ Wih_2,  const float* __restrict__ Whh_2,
    const float* __restrict__ bih_2,  const float* __restrict__ bhh_2,
    const float* __restrict__ W_lin,  const float* __restrict__ b_lin,
    float* __restrict__ out) {
  extern __shared__ char smem[];
  _Float16* Bc1  = (_Float16*)(smem + OFF_BC1);
  _Float16* Bc2  = (_Float16*)(smem + OFF_BC2);
  _Float16* Bs1  = (_Float16*)(smem + OFF_BS1);
  _Float16* Bs2  = (_Float16*)(smem + OFF_BS2);
  _Float16* Bpro = (_Float16*)(smem + OFF_BPRO);
  _Float16* Ac1  = (_Float16*)(smem + OFF_AC1);
  _Float16* Ac2  = (_Float16*)(smem + OFF_AC2);
  _Float16* As1  = (_Float16*)(smem + OFF_AS1);
  _Float16* As2  = (_Float16*)(smem + OFF_AS2);
  _Float16* Apro = (_Float16*)(smem + OFF_APRO);
  float* g      = (float*)(smem + OFF_G);
  float* h2f    = (float*)(smem + OFF_H2F);
  float* xc1    = (float*)(smem + OFF_XC1);
  float* xc2    = (float*)(smem + OFF_XC2);
  float* WiC1   = (float*)(smem + OFF_WIC1);
  float* WiC2   = (float*)(smem + OFF_WIC2);
  float* biasC1 = (float*)(smem + OFF_BI_C1);
  float* biasC2 = (float*)(smem + OFF_BI_C2);
  float* biasS1 = (float*)(smem + OFF_BI_S1);
  float* biasS2 = (float*)(smem + OFF_BI_S2);
  float* bproL  = (float*)(smem + OFF_BPROB);
  float* WlinL  = (float*)(smem + OFF_WLIN);
  float* blinL  = (float*)(smem + OFF_BLIN);

  const int tid  = threadIdx.x;
  const int wave = tid >> 5, lane = tid & 31, hh = lane >> 4, m = lane & 15;
  const int mm = tid >> 4, n0 = (tid & 15) * 4;   // 4 gate cells per thread
  const int b0 = blockIdx.x * BT;

  // ---- one-time init: swizzle weights into WMMA B-fragment layout in LDS ----
  // frag elem idx = ((kt*NT + nt)*32 + lane)*16 + i ; K = kt*32 + (lane>>4)*16 + i ; N = nt*16 + (lane&15)
  for (int idx = tid; idx < 2*16*32*16; idx += NTH) {          // c1 / c2 (K=64)
    int i = idx & 15, ln = (idx >> 4) & 31, nt = (idx >> 9) & 15, kt = idx >> 13;
    int K = kt*32 + (ln >> 4)*16 + i, N = nt*16 + (ln & 15);
    Bc1[idx] = (_Float16)Whh_c1[N*64 + K];
    Bc2[idx] = (_Float16)Whh_c2[N*64 + K];
  }
  for (int idx = tid; idx < 4*16*32*16; idx += NTH) {          // s1 / s2 (K=128)
    int i = idx & 15, ln = (idx >> 4) & 31, nt = (idx >> 9) & 15, kt = idx >> 13;
    int K = kt*32 + (ln >> 4)*16 + i, N = nt*16 + (ln & 15);
    float v1 = (K < 64) ? Whh_1[N*64 + K] : ((K < 99) ? Wih_1[N*35 + (K - 64)] : 0.f);
    float v2 = (K < 64) ? Whh_2[N*64 + K] : Wih_2[N*64 + (K - 64)];
    Bs1[idx] = (_Float16)v1;
    Bs2[idx] = (_Float16)v2;
  }
  for (int idx = tid; idx < 4*2*32*16; idx += NTH) {           // projection (K=128, NT=2)
    int i = idx & 15, ln = (idx >> 4) & 31, nt = (idx >> 9) & 1, kt = idx >> 10;
    int K = kt*32 + (ln >> 4)*16 + i, N = nt*16 + (ln & 15);
    Bpro[idx] = (_Float16)W_pro[N*128 + K];
  }
  for (int i = tid; i < 256; i += NTH) {
    biasC1[i] = bih_c1[i] + bhh_c1[i];
    biasC2[i] = bih_c2[i] + bhh_c2[i];
    biasS1[i] = bih_1[i]  + bhh_1[i];
    biasS2[i] = bih_2[i]  + bhh_2[i];
    WiC2[i]   = Wih_c2[i];
  }
  for (int i = tid; i < 1024; i += NTH) WiC1[i] = Wih_c1[i];
  if (tid < 32) bproL[tid] = b_pro[tid];
  if (tid < 64) WlinL[tid] = W_lin[tid];
  if (tid == 0) blinL[0] = b_lin[0];
  { // zero all A buffers (h0 = c0 = 0, plus permanent K-pads)
    _Float16* Az = (_Float16*)(smem + OFF_AC1);
    for (int i = tid; i < 8192; i += NTH) Az[i] = (_Float16)0.f;
  }
  float cC1[4], cC2[4], cS1[4], cS2[4];
  #pragma unroll
  for (int q = 0; q < 4; ++q) { cC1[q] = 0.f; cC2[q] = 0.f; cS1[q] = 0.f; cS2[q] = 0.f; }

  // Per-thread staging assignment with hoisted 64-bit global base pointers
  // (threads 0..127 stage 16 rows x 8 scalars per step).
  const int sr = tid >> 3, sj = tid & 7;
  const float* stage_src = nullptr;   // per-step: stage_src[t * stage_mul]
  int stage_mul = 0;
  _Float16* stage_dst_h = nullptr;    // f16 dest (raw inputs into As1)
  float* stage_dst_f = nullptr;       // f32 dest (ctx inputs)
  if (tid < 128) {
    long b = (long)(b0 + sr);
    if (sj < 3) {
      stage_src = inputs + b * T_LEN * 3 + sj;  stage_mul = 3;
      stage_dst_h = &As1[sr * 128 + 64 + sj];
    } else if (sj < 7) {
      stage_src = ctx1 + b * T_LEN * 4 + (sj - 3);  stage_mul = 4;
      stage_dst_f = &xc1[sr * 4 + (sj - 3)];
    } else {
      stage_src = ctx2 + b * T_LEN;  stage_mul = 1;
      stage_dst_f = &xc2[sr];
    }
  }
  float* outp = out + (long)(b0 + tid) * T_LEN;  // head rows (tid<16)
  __syncthreads();

  // ---- persistent time loop ----
  for (int t = 0; t < T_LEN; ++t) {
    if (tid < 128) {                      // stage this step's inputs
      float v = stage_src[(long)t * stage_mul];
      if (stage_dst_h) *stage_dst_h = (_Float16)v;
      else             *stage_dst_f = v;
    }
    __syncthreads();

    // context LSTM 1: recurrent K=64 in WMMA; 4-dim input folded into VALU phase
    gate_gemm<2>(Ac1, 64, Bc1, g, wave, lane, hh, m);
    __syncthreads();
    lstm_elem<4>(g, biasC1, WiC1, &xc1[mm*4], cC1, Ac1, 64, 0, Apro, 128, 0, nullptr, mm, n0);
    __syncthreads();

    // context LSTM 2 (1-dim input)
    gate_gemm<2>(Ac2, 64, Bc2, g, wave, lane, hh, m);
    __syncthreads();
    lstm_elem<1>(g, biasC2, WiC2, &xc2[mm], cC2, Ac2, 64, 0, Apro, 128, 64, nullptr, mm, n0);
    __syncthreads();

    // projection 128 -> 32 (2 N-tiles on waves 0,1), result -> As1 cols 67..98 (f16)
    if (wave < 2) {
      v8f acc = {};
      #pragma unroll
      for (int kt = 0; kt < 4; ++kt) {
        v16h a = load_a_frag(Apro, 128, m, hh, kt * 32);
        v16h b = *(const v16h*)(Bpro + (((kt*2 + wave)*32) + lane) * 16);
        acc = __builtin_amdgcn_wmma_f32_16x16x32_f16(false, a, false, b, (short)0, acc, false, false);
      }
      float bb = bproL[wave*16 + m];
      #pragma unroll
      for (int r = 0; r < 8; ++r)
        As1[(r + hh*8)*128 + 67 + wave*16 + m] = (_Float16)(acc[r] + bb);
    }
    __syncthreads();

    // stacked LSTM 1: A = [h1 | raw3 | proj32 | pad], K=128
    gate_gemm<4>(As1, 128, Bs1, g, wave, lane, hh, m);
    __syncthreads();
    lstm_elem<0>(g, biasS1, nullptr, nullptr, cS1, As1, 128, 0, As2, 128, 64, nullptr, mm, n0);
    __syncthreads();

    // stacked LSTM 2: A = [h2 | h1], K=128
    gate_gemm<4>(As2, 128, Bs2, g, wave, lane, hh, m);
    __syncthreads();
    lstm_elem<0>(g, biasS2, nullptr, nullptr, cS2, As2, 128, 0, nullptr, 0, 0, h2f, mm, n0);
    __syncthreads();

    // head: 64 -> 1 linear + SELU
    if (tid < 16) {
      float acc = blinL[0];
      #pragma unroll 8
      for (int k = 0; k < 64; ++k) acc += h2f[tid*64 + k] * WlinL[k];
      float r = acc > 0.f ? acc : 1.6732632423543772f * (__expf(acc) - 1.f);
      outp[t] = 1.0507009873554805f * r;
    }
    __syncthreads();
  }
}

extern "C" void kernel_launch(void* const* d_in, const int* in_sizes, int n_in,
                              void* d_out, int out_size, void* d_ws, size_t ws_size,
                              hipStream_t stream) {
  (void)in_sizes; (void)n_in; (void)out_size; (void)d_ws; (void)ws_size;
  const float* inputs = (const float*)d_in[0];
  const float* ctx1   = (const float*)d_in[1];
  const float* ctx2   = (const float*)d_in[2];
  const float* Wih_c1 = (const float*)d_in[3];
  const float* Whh_c1 = (const float*)d_in[4];
  const float* bih_c1 = (const float*)d_in[5];
  const float* bhh_c1 = (const float*)d_in[6];
  const float* Wih_c2 = (const float*)d_in[7];
  const float* Whh_c2 = (const float*)d_in[8];
  const float* bih_c2 = (const float*)d_in[9];
  const float* bhh_c2 = (const float*)d_in[10];
  const float* W_pro  = (const float*)d_in[11];
  const float* b_pro  = (const float*)d_in[12];
  const float* Wih_1  = (const float*)d_in[13];
  const float* Whh_1  = (const float*)d_in[14];
  const float* bih_1  = (const float*)d_in[15];
  const float* bhh_1  = (const float*)d_in[16];
  const float* Wih_2  = (const float*)d_in[17];
  const float* Whh_2  = (const float*)d_in[18];
  const float* bih_2  = (const float*)d_in[19];
  const float* bhh_2  = (const float*)d_in[20];
  const float* W_lin  = (const float*)d_in[21];
  const float* b_lin  = (const float*)d_in[22];
  float* out = (float*)d_out;

  // Opt in to >64KB dynamic LDS (gfx1250 WGP: 320KB). Idempotent, capture-safe.
  (void)hipFuncSetAttribute((const void*)fused_lstm_stack_kernel,
                            hipFuncAttributeMaxDynamicSharedMemorySize, SMEM_BYTES);

  dim3 grid(B_TOT / BT), block(NTH);
  fused_lstm_stack_kernel<<<grid, block, SMEM_BYTES, stream>>>(
      inputs, ctx1, ctx2,
      Wih_c1, Whh_c1, bih_c1, bhh_c1,
      Wih_c2, Whh_c2, bih_c2, bhh_c2,
      W_pro, b_pro,
      Wih_1, Whh_1, bih_1, bhh_1,
      Wih_2, Whh_2, bih_2, bhh_2,
      W_lin, b_lin, out);
}
